// MultimodalGAT_70875550318712
// MI455X (gfx1250) — compile-verified, matrix operand order checked by reading, not run on hardware
//
#include <hip/hip_runtime.h>
#include <hip/hip_bf16.h>
#include <stdint.h>

#define NEG_SLOPE 0.2f
#define BN_EPS 1e-5f

typedef __attribute__((ext_vector_type(16))) __bf16 v16bf;
typedef __attribute__((ext_vector_type(8)))  float  v8f;

__device__ __forceinline__ unsigned short f2bf_bits(float f) {
  unsigned u = __float_as_uint(f);
  unsigned r = u + 0x7FFFu + ((u >> 16) & 1u);   // round-to-nearest-even
  return (unsigned short)(r >> 16);
}
__device__ __forceinline__ void atomicMaxF(float* addr, float val) {
  int old = __float_as_int(*addr);
  while (__int_as_float(old) < val) {
    int assumed = old;
    old = atomicCAS((int*)addr, assumed, __float_as_int(val));
    if (old == assumed) break;
  }
}

// ---------------- weight fp32[K,Nc] -> bf16 transposed Wt[Nc,K] ----------------
__global__ void k_cvt_tr_bf16(const float* __restrict__ in, unsigned short* __restrict__ out,
                              int K, int Nc) {
  int i = blockIdx.x * blockDim.x + threadIdx.x;
  if (i >= K * Nc) return;
  int n = i / K, k = i - n * K;
  out[i] = f2bf_bits(in[(size_t)k * Nc + n]);   // write-coalesced
}

// ---------------- fills ----------------
__global__ void k_fill(float* __restrict__ p, float v, int n) {
  int i = blockIdx.x * blockDim.x + threadIdx.x;
  if (i < n) p[i] = v;
}
__global__ void k_fill_bias(float* __restrict__ p, const float* __restrict__ b, int total, int c) {
  int i = blockIdx.x * blockDim.x + threadIdx.x;
  if (i < total) p[i] = b[i % c];
}

// ---------------- bf16 WMMA GEMM: C[M,Nc] = A[M,K] * W[K,Nc], W given transposed ----------------
// block = 128 threads (4 waves); block tile = 32 rows x 256 cols; wave tile = 32x64
// (2 row stripes x 4 col tiles = 8 WMMA accumulators; B fragments reused across row stripes).
// LDS double-buffered; fragment reads ds_load_b128; staging global_load_b128 + prefetch.
#define TCOLS 256
#define TROWS 32
#define LSTR  40   // padded LDS stride in ushorts (80B, 16B aligned, bank-friendly)
__global__ __launch_bounds__(128)
void k_gemm_bf16(const float* __restrict__ A, const unsigned short* __restrict__ Wt,
                 float* __restrict__ C, int M, int K, int Nc) {
  __shared__ unsigned short sA[2][TROWS * LSTR];
  __shared__ unsigned short sB[2][TCOLS * LSTR];
  const int row0 = blockIdx.x * TROWS;
  const int col0 = blockIdx.y * TCOLS;
  const int t    = threadIdx.x;
  const int wave = t >> 5;
  const int lane = t & 31;
  const int g    = lane >> 4;   // lane group (0/1)
  const int r    = lane & 15;   // row (A) / col (B,C,D) within 16x16 tile

  v8f acc[2][4] = {};

  auto stage = [&](int buf, int k0) {
    // A tile: 32x32 fp32 -> bf16 ; 256 float4 chunks over 128 threads
#pragma unroll
    for (int it = 0; it < 2; ++it) {
      int chunk = it * 128 + t;
      int rr = chunk >> 3, kk = (chunk & 7) * 4;
      const float4 av = *(const float4*)&A[(size_t)(row0 + rr) * K + (k0 + kk)];
      uint2 p;
      p.x = (unsigned)f2bf_bits(av.x) | ((unsigned)f2bf_bits(av.y) << 16);
      p.y = (unsigned)f2bf_bits(av.z) | ((unsigned)f2bf_bits(av.w) << 16);
      *(uint2*)&sA[buf][rr * LSTR + kk] = p;
    }
    // W tile: 256 cols x 32 k (bf16, K-contiguous in Wt) ; uint4 = 8 bf16 per op
#pragma unroll
    for (int it = 0; it < 8; ++it) {
      int idx = it * 128 + t;          // 1024 16B chunks
      int cc  = idx >> 2;              // 0..255
      int kk  = (idx & 3) * 8;         // 0,8,16,24
      uint4 w = *(const uint4*)&Wt[(size_t)(col0 + cc) * K + (k0 + kk)];
      *(uint4*)&sB[buf][cc * LSTR + kk] = w;
    }
  };

  stage(0, 0);
  __syncthreads();

  const int nsteps = K >> 5;
  for (int s = 0; s < nsteps; ++s) {
    const int cur = s & 1;
    if (s + 2 < nsteps) {   // prefetch HBM lines two tiles ahead into GL2
      __builtin_prefetch(&A[(size_t)(row0 + (t >> 3)) * K + ((s + 2) << 5) + (t & 7) * 4], 0, 1);
      __builtin_prefetch(&Wt[(size_t)(col0 + (t << 1)) * K + ((s + 2) << 5)], 0, 1);
    }
    if (s + 1 < nsteps) stage(cur ^ 1, (s + 1) << 5);

    union FR { v16bf v; uint4 q[2]; };
    FR fa0, fa1;
    fa0.q[0] = *(const uint4*)&sA[cur][r * LSTR + g * 8];              // rows 0..15
    fa0.q[1] = *(const uint4*)&sA[cur][r * LSTR + 16 + g * 8];
    fa1.q[0] = *(const uint4*)&sA[cur][(16 + r) * LSTR + g * 8];       // rows 16..31
    fa1.q[1] = *(const uint4*)&sA[cur][(16 + r) * LSTR + 16 + g * 8];
#pragma unroll
    for (int tt = 0; tt < 4; ++tt) {
      int c = wave * 64 + tt * 16 + r;   // lane holds column c
      FR fb;
      fb.q[0] = *(const uint4*)&sB[cur][c * LSTR + g * 8];
      fb.q[1] = *(const uint4*)&sB[cur][c * LSTR + 16 + g * 8];
      acc[0][tt] = __builtin_amdgcn_wmma_f32_16x16x32_bf16(
          false, fa0.v, false, fb.v, (short)0, acc[0][tt], false, false);
      acc[1][tt] = __builtin_amdgcn_wmma_f32_16x16x32_bf16(
          false, fa1.v, false, fb.v, (short)0, acc[1][tt], false, false);
    }
    __syncthreads();
  }

  // store: acc[rs][tt] element v -> (M = row0 + rs*16 + g*8 + v, N = col0 + wave*64 + tt*16 + r)
#pragma unroll
  for (int rs = 0; rs < 2; ++rs) {
#pragma unroll
    for (int tt = 0; tt < 4; ++tt) {
#pragma unroll
      for (int v = 0; v < 8; ++v) {
        int m = rs * 16 + g * 8 + v;
        int c = wave * 64 + tt * 16 + r;
        C[(size_t)(row0 + m) * Nc + (col0 + c)] = acc[rs][tt][v];
      }
    }
  }
}

// ---------------- edge score: wave per (edge, head), float4 gathers ----------------
__global__ __launch_bounds__(256)
void k_edge_score(const float* __restrict__ xl, const float* __restrict__ xr,
                  const float* __restrict__ ea, const float* __restrict__ we,
                  const float* __restrict__ att, const int* __restrict__ ei,
                  float* __restrict__ sc, float* __restrict__ nmax,
                  int E, int H, int C) {
  int pair = blockIdx.x * 8 + (threadIdx.x >> 5);
  int lane = threadIdx.x & 31;
  if (pair >= E * H) return;
  int e = pair / H, h = pair - e * H;
  int src = ei[e], dst = ei[E + e];
  float a = ea[e];
  int HC = H * C;
  const float* pl = xl + (size_t)src * HC + h * C;
  const float* pr = xr + (size_t)dst * HC + h * C;
  const float* pw = we + h * C;
  const float* pa = att + h * C;
  float sum = 0.f;
  for (int c = lane * 4; c < C; c += 128) {
    float4 l4 = *(const float4*)&pl[c];
    float4 r4 = *(const float4*)&pr[c];
    float4 w4 = *(const float4*)&pw[c];
    float4 a4 = *(const float4*)&pa[c];
    float m0 = l4.x + r4.x + a * w4.x; m0 = (m0 > 0.f) ? m0 : NEG_SLOPE * m0;
    float m1 = l4.y + r4.y + a * w4.y; m1 = (m1 > 0.f) ? m1 : NEG_SLOPE * m1;
    float m2 = l4.z + r4.z + a * w4.z; m2 = (m2 > 0.f) ? m2 : NEG_SLOPE * m2;
    float m3 = l4.w + r4.w + a * w4.w; m3 = (m3 > 0.f) ? m3 : NEG_SLOPE * m3;
    sum += m0 * a4.x + m1 * a4.y + m2 * a4.z + m3 * a4.w;
  }
  for (int off = 16; off > 0; off >>= 1) sum += __shfl_xor(sum, off, 32);
  if (lane == 0) {
    sc[pair] = sum;
    atomicMaxF(&nmax[dst * H + h], sum);
  }
}

// ---------------- exp + segment sum ----------------
__global__ void k_edge_exp(float* __restrict__ sc, const float* __restrict__ nmax,
                           float* __restrict__ nsum, const int* __restrict__ ei,
                           int E, int H) {
  int i = blockIdx.x * blockDim.x + threadIdx.x;
  if (i >= E * H) return;
  int e = i / H, h = i - e * H;
  int dst = ei[E + e];
  float v = expf(sc[i] - nmax[dst * H + h]);
  sc[i] = v;
  atomicAdd(&nsum[dst * H + h], v);
}

// ---------------- weighted scatter-add: wave per (edge, head) ----------------
__global__ __launch_bounds__(256)
void k_edge_scatter(const float* __restrict__ xl, const float* __restrict__ sc,
                    const float* __restrict__ nsum, const int* __restrict__ ei,
                    float* __restrict__ out, int E, int H, int C) {
  int pair = blockIdx.x * 8 + (threadIdx.x >> 5);
  int lane = threadIdx.x & 31;
  if (pair >= E * H) return;
  int e = pair / H, h = pair - e * H;
  int src = ei[e], dst = ei[E + e];
  float alpha = sc[pair] / (nsum[dst * H + h] + 1e-16f);
  int HC = H * C;
  const float* pl = xl + (size_t)src * HC + h * C;
  float* po = out + (size_t)dst * HC + h * C;
  for (int c = lane * 4; c < C; c += 128) {
    float4 l4 = *(const float4*)&pl[c];
    atomicAdd(&po[c + 0], alpha * l4.x);
    atomicAdd(&po[c + 1], alpha * l4.y);
    atomicAdd(&po[c + 2], alpha * l4.z);
    atomicAdd(&po[c + 3], alpha * l4.w);
  }
}

// ---------------- BN stats: per-block register partials + atomics ----------------
__global__ __launch_bounds__(256)
void k_bn_stats(const float* __restrict__ x, float* __restrict__ sums,
                float* __restrict__ sqs, int n, int c, int rpb) {
  int row0 = blockIdx.x * rpb;
  int row1 = min(row0 + rpb, n);
  for (int ch = threadIdx.x; ch < c; ch += blockDim.x) {
    float s = 0.f, q = 0.f;
    for (int r = row0; r < row1; ++r) {
      float v = x[(size_t)r * c + ch];
      s += v; q += v * v;
    }
    atomicAdd(&sums[ch], s);
    atomicAdd(&sqs[ch], q);
  }
}
__global__ void k_bn_apply(float* __restrict__ x, const float* __restrict__ sums,
                           const float* __restrict__ sqs, const float* __restrict__ g,
                           const float* __restrict__ b, int n, int c) {
  int i = blockIdx.x * blockDim.x + threadIdx.x;
  if (i >= n * c) return;
  int ch = i % c;
  float inv = 1.f / (float)n;
  float mean = sums[ch] * inv;
  float var  = sqs[ch] * inv - mean * mean;
  float y = (x[i] - mean) * rsqrtf(var + BN_EPS) * g[ch] + b[ch];
  x[i] = (y > 0.f) ? y : (expf(y) - 1.f);   // ELU fused
}

// ---------------- mean over 4 heads + bias ----------------
__global__ void k_head_mean(const float* __restrict__ o, const float* __restrict__ bias,
                            float* __restrict__ h1, int n) {
  int i = blockIdx.x * blockDim.x + threadIdx.x;
  if (i >= n * 64) return;
  int nd = i >> 6, c = i & 63;
  size_t base = (size_t)nd * 256;
  h1[i] = 0.25f * (o[base + c] + o[base + 64 + c] + o[base + 128 + c] + o[base + 192 + c]) + bias[c];
}

// ---------------- global mean pool ----------------
__global__ void k_pool(const float* __restrict__ h, const int* __restrict__ batch,
                       float* __restrict__ pool, int n) {
  int i = blockIdx.x * blockDim.x + threadIdx.x;
  if (i >= n * 64) return;
  int nd = i >> 6, c = i & 63;
  atomicAdd(&pool[batch[nd] * 64 + c], h[i]);
}
__global__ void k_count(const int* __restrict__ batch, float* __restrict__ cnt, int n) {
  int i = blockIdx.x * blockDim.x + threadIdx.x;
  if (i < n) atomicAdd(&cnt[batch[i]], 1.f);
}
__global__ void k_embdiv(const float* __restrict__ pool, const float* __restrict__ cnt,
                         float* __restrict__ emb, int B) {
  int i = blockIdx.x * blockDim.x + threadIdx.x;
  if (i < B * 64) emb[i] = pool[i] / fmaxf(cnt[i >> 6], 1.f);
}

// ---------------- classifier: single block, thread owns a full column ----------------
__global__ __launch_bounds__(256)
void k_classifier(const float* __restrict__ emb,
                  const float* __restrict__ w1, const float* __restrict__ b1,
                  const float* __restrict__ g1, const float* __restrict__ be1,
                  const float* __restrict__ w2, const float* __restrict__ b2,
                  const float* __restrict__ g2, const float* __restrict__ be2,
                  const float* __restrict__ w3, const float* __restrict__ b3,
                  float* __restrict__ z1, float* __restrict__ z2,
                  float* __restrict__ outp) {
  const int t = threadIdx.x;
  // stage 1: 64 -> 256, BN over 64 rows + ELU
  {
    float v[64];
    for (int r = 0; r < 64; ++r) {
      float s = b1[t];
      for (int k = 0; k < 64; ++k) s += emb[r * 64 + k] * w1[k * 256 + t];
      v[r] = s;
    }
    float m = 0.f, q = 0.f;
    for (int r = 0; r < 64; ++r) { m += v[r]; q += v[r] * v[r]; }
    m *= (1.f / 64.f); q = q * (1.f / 64.f) - m * m;
    float sc = rsqrtf(q + BN_EPS) * g1[t];
    float sh = be1[t] - m * sc;
    for (int r = 0; r < 64; ++r) {
      float y = v[r] * sc + sh;
      z1[r * 256 + t] = (y > 0.f) ? y : (expf(y) - 1.f);
    }
  }
  __syncthreads();
  // stage 2: 256 -> 128
  if (t < 128) {
    float v[64];
    for (int r = 0; r < 64; ++r) {
      float s = b2[t];
      for (int k = 0; k < 256; ++k) s += z1[r * 256 + k] * w2[k * 128 + t];
      v[r] = s;
    }
    float m = 0.f, q = 0.f;
    for (int r = 0; r < 64; ++r) { m += v[r]; q += v[r] * v[r]; }
    m *= (1.f / 64.f); q = q * (1.f / 64.f) - m * m;
    float sc = rsqrtf(q + BN_EPS) * g2[t];
    float sh = be2[t] - m * sc;
    for (int r = 0; r < 64; ++r) {
      float y = v[r] * sc + sh;
      z2[r * 128 + t] = (y > 0.f) ? y : (expf(y) - 1.f);
    }
  }
  __syncthreads();
  // stage 3: 128 -> 2 logits
  if (t < 2) {
    for (int r = 0; r < 64; ++r) {
      float s = b3[t];
      for (int k = 0; k < 128; ++k) s += z2[r * 128 + k] * w3[k * 2 + t];
      outp[r * 2 + t] = s;
    }
  }
  // emb appended after logits
  for (int i = t; i < 64 * 64; i += 256) outp[128 + i] = emb[i];
}

extern "C" void kernel_launch(void* const* d_in, const int* in_sizes, int n_in,
                              void* d_out, int out_size, void* d_ws, size_t ws_size,
                              hipStream_t stream) {
  (void)n_in; (void)out_size; (void)ws_size;
  const float* x      = (const float*)d_in[0];
  const int*   ei     = (const int*)d_in[1];
  const float* eattr  = (const float*)d_in[2];
  const int*   batch  = (const int*)d_in[3];
  const float* g0_wl  = (const float*)d_in[4];
  const float* g0_wr  = (const float*)d_in[5];
  const float* g0_we  = (const float*)d_in[6];
  const float* g0_att = (const float*)d_in[7];
  const float* g0_bias= (const float*)d_in[8];
  const float* bn0_g  = (const float*)d_in[9];
  const float* bn0_b  = (const float*)d_in[10];
  const float* g1_wl  = (const float*)d_in[11];
  const float* g1_wr  = (const float*)d_in[12];
  const float* g1_we  = (const float*)d_in[13];
  const float* g1_att = (const float*)d_in[14];
  const float* g1_bias= (const float*)d_in[15];
  const float* bn1_g  = (const float*)d_in[16];
  const float* bn1_b  = (const float*)d_in[17];
  const float* c_w1   = (const float*)d_in[18];
  const float* c_b1   = (const float*)d_in[19];
  const float* cbn1_g = (const float*)d_in[20];
  const float* cbn1_b = (const float*)d_in[21];
  const float* c_w2   = (const float*)d_in[22];
  const float* c_b2   = (const float*)d_in[23];
  const float* cbn2_g = (const float*)d_in[24];
  const float* cbn2_b = (const float*)d_in[25];
  const float* c_w3   = (const float*)d_in[26];
  const float* c_b3   = (const float*)d_in[27];

  const int N = in_sizes[0] / 256;   // 20000
  const int E = in_sizes[2];         // 320000
  const int B = 64;

  // ---- workspace layout ----
  char* ws = (char*)d_ws;
  size_t off = 0;
  auto alloc = [&](size_t bytes) -> void* {
    void* p = ws + off;
    off += (bytes + 255) & ~(size_t)255;
    return p;
  };
  unsigned short* w0l = (unsigned short*)alloc((size_t)256 * 512 * 2);  // transposed [512,256]
  unsigned short* w0r = (unsigned short*)alloc((size_t)256 * 512 * 2);
  unsigned short* w1l = (unsigned short*)alloc((size_t)512 * 256 * 2);  // transposed [256,512]
  unsigned short* w1r = (unsigned short*)alloc((size_t)512 * 256 * 2);
  float* XL   = (float*)alloc((size_t)N * 512 * 4);
  float* XR   = (float*)alloc((size_t)N * 512 * 4);
  float* OUT  = (float*)alloc((size_t)N * 512 * 4);
  float* SCb  = (float*)alloc((size_t)E * 4 * 4);
  float* H1B  = (float*)alloc((size_t)N * 64 * 4);
  float* NMAX = (float*)alloc((size_t)N * 4 * 4);
  float* NSUM = (float*)alloc((size_t)N * 4 * 4);
  float* SSUM = (float*)alloc(512 * 4);
  float* SSQ  = (float*)alloc(512 * 4);
  float* POOL = (float*)alloc((size_t)B * 64 * 4);
  float* CNT  = (float*)alloc((size_t)B * 4);
  float* Z1   = (float*)alloc((size_t)64 * 256 * 4);
  float* Z2   = (float*)alloc((size_t)64 * 128 * 4);
  float* EMB  = (float*)alloc((size_t)64 * 64 * 4);

  const int WN = 256 * 512;   // all four weight matrices have 131072 elements
  k_cvt_tr_bf16<<<(WN + 255) / 256, 256, 0, stream>>>(g0_wl, w0l, 256, 512);
  k_cvt_tr_bf16<<<(WN + 255) / 256, 256, 0, stream>>>(g0_wr, w0r, 256, 512);
  k_cvt_tr_bf16<<<(WN + 255) / 256, 256, 0, stream>>>(g1_wl, w1l, 512, 256);
  k_cvt_tr_bf16<<<(WN + 255) / 256, 256, 0, stream>>>(g1_wr, w1r, 512, 256);

  // ---- layer 0: GEMMs (bf16 WMMA) ----
  dim3 g0grid(N / TROWS, 512 / TCOLS);
  k_gemm_bf16<<<g0grid, 128, 0, stream>>>(x, w0l, XL, N, 256, 512);
  k_gemm_bf16<<<g0grid, 128, 0, stream>>>(x, w0r, XR, N, 256, 512);

  // ---- layer 0: edge phase ----
  k_fill_bias<<<(N * 512 + 255) / 256, 256, 0, stream>>>(OUT, g0_bias, N * 512, 512);
  k_fill<<<(N * 4 + 255) / 256, 256, 0, stream>>>(NMAX, -1e30f, N * 4);
  k_fill<<<(N * 4 + 255) / 256, 256, 0, stream>>>(NSUM, 0.f, N * 4);
  int pairs = E * 4;
  k_edge_score  <<<(pairs + 7) / 8,     256, 0, stream>>>(XL, XR, eattr, g0_we, g0_att, ei, SCb, NMAX, E, 4, 128);
  k_edge_exp    <<<(pairs + 255) / 256, 256, 0, stream>>>(SCb, NMAX, NSUM, ei, E, 4);
  k_edge_scatter<<<(pairs + 7) / 8,     256, 0, stream>>>(XL, SCb, NSUM, ei, OUT, E, 4, 128);

  // ---- BN0 + ELU in place ----
  k_fill<<<2, 256, 0, stream>>>(SSUM, 0.f, 512);
  k_fill<<<2, 256, 0, stream>>>(SSQ,  0.f, 512);
  k_bn_stats<<<(N + 63) / 64, 256, 0, stream>>>(OUT, SSUM, SSQ, N, 512, 64);
  k_bn_apply<<<(N * 512 + 255) / 256, 256, 0, stream>>>(OUT, SSUM, SSQ, bn0_g, bn0_b, N, 512);

  // ---- layer 1: GEMMs ----
  dim3 g1grid(N / TROWS, 256 / TCOLS);
  k_gemm_bf16<<<g1grid, 128, 0, stream>>>(OUT, w1l, XL, N, 512, 256);
  k_gemm_bf16<<<g1grid, 128, 0, stream>>>(OUT, w1r, XR, N, 512, 256);

  // ---- layer 1: edge phase (OUT reused as [N,256] accumulator) ----
  k_fill<<<(N * 256 + 255) / 256, 256, 0, stream>>>(OUT, 0.f, N * 256);
  k_fill<<<(N * 4 + 255) / 256, 256, 0, stream>>>(NMAX, -1e30f, N * 4);
  k_fill<<<(N * 4 + 255) / 256, 256, 0, stream>>>(NSUM, 0.f, N * 4);
  k_edge_score  <<<(pairs + 7) / 8,     256, 0, stream>>>(XL, XR, eattr, g1_we, g1_att, ei, SCb, NMAX, E, 4, 64);
  k_edge_exp    <<<(pairs + 255) / 256, 256, 0, stream>>>(SCb, NMAX, NSUM, ei, E, 4);
  k_edge_scatter<<<(pairs + 7) / 8,     256, 0, stream>>>(XL, SCb, NSUM, ei, OUT, E, 4, 64);

  // ---- head mean + bias, BN1 + ELU ----
  k_head_mean<<<(N * 64 + 255) / 256, 256, 0, stream>>>(OUT, g1_bias, H1B, N);
  k_fill<<<1, 256, 0, stream>>>(SSUM, 0.f, 64);
  k_fill<<<1, 256, 0, stream>>>(SSQ,  0.f, 64);
  k_bn_stats<<<(N + 63) / 64, 256, 0, stream>>>(H1B, SSUM, SSQ, N, 64, 64);
  k_bn_apply<<<(N * 64 + 255) / 256, 256, 0, stream>>>(H1B, SSUM, SSQ, bn1_g, bn1_b, N, 64);

  // ---- global mean pool ----
  k_fill<<<(B * 64 + 255) / 256, 256, 0, stream>>>(POOL, 0.f, B * 64);
  k_fill<<<1, 256, 0, stream>>>(CNT, 0.f, B);
  k_pool <<<(N * 64 + 255) / 256, 256, 0, stream>>>(H1B, batch, POOL, N);
  k_count<<<(N + 255) / 256, 256, 0, stream>>>(batch, CNT, N);
  k_embdiv<<<(B * 64 + 255) / 256, 256, 0, stream>>>(POOL, CNT, EMB, B);

  // ---- classifier -> d_out = [logits(64x2), emb(64x64)] ----
  k_classifier<<<1, 256, 0, stream>>>(EMB, c_w1, c_b1, cbn1_g, cbn1_b,
                                      c_w2, c_b2, cbn2_g, cbn2_b,
                                      c_w3, c_b3, Z1, Z2, (float*)d_out);
}
